// Decoder_78606491452121
// MI455X (gfx1250) — matrix operand order repaired and checked by reading, not compiled, not run on hardware
//
#include <hip/hip_runtime.h>

#define B_   64
#define L_   64
#define T_   15
#define V_   32000
#define E_   512
#define H_   1024
#define ENC_ 1024
#define ATT_ 512
#define S_   128
#define XDIM (E_ + S_ + ENC_)   /* 1664 */

typedef __attribute__((ext_vector_type(16))) __bf16        v16bf;
typedef __attribute__((ext_vector_type(8)))  float         v8f;
typedef __attribute__((ext_vector_type(8)))  unsigned int  v8u;

// fp32 -> bf16 round-to-nearest-even (used OFF the GEMM hot path only)
static __device__ __forceinline__ __bf16 f2bf(float f) {
  unsigned int u = __builtin_bit_cast(unsigned int, f);
  unsigned int r = u + 0x7FFFu + ((u >> 16) & 1u);
  return __builtin_bit_cast(__bf16, (unsigned short)(r >> 16));
}

// pack bf16(lo),bf16(hi) into one dword with a single v_perm_b32 (truncation)
static __device__ __forceinline__ unsigned int pack_hi(unsigned int lo, unsigned int hi) {
  return __builtin_amdgcn_perm(hi, lo, 0x07060302u);
}

// =============================================================================
// GEMM #1: A bf16 [M,K], W bf16 [N,K]  ->  C = act(A @ W^T + bias)
// One 16x16 C tile per wave; K in steps of 32 via v_wmma_f32_16x16x32_bf16.
// grid=(M/16, N/64), block=128 (4 waves). All K multiples of 64, N of 64.
// A/B fragments use contiguous k = 16*(lane/16) + s, same for both; f32 C 16x16 as in ISA 7.12.2.
// =============================================================================
__global__ void gemm_bb(const __bf16* __restrict__ A, const __bf16* __restrict__ W,
                        const float* __restrict__ bias, float* __restrict__ Cf,
                        __bf16* __restrict__ Ch, int K, int ldc, int act) {
  const int lane    = threadIdx.x & 31;
  const int wave    = threadIdx.x >> 5;
  const int rowbase = blockIdx.x * 16;
  const int colbase = (blockIdx.y * 4 + wave) * 16;
  const int arow    = rowbase + (lane & 15);
  const int bcol    = colbase + (lane & 15);
  const int kA      = (lane < 16) ? 0 : 16;
  const int kB      = (lane < 16) ? 0 : 16;
  const __bf16* __restrict__ Ap = A + (size_t)arow * K;
  const __bf16* __restrict__ Wp = W + (size_t)bcol * K;

  v8f acc = {};
#pragma unroll 2
  for (int kk = 0; kk < K; kk += 32) {
    __builtin_prefetch((const char*)(Wp + kk) + 4096, 0, 1);   // global_prefetch
    uint4 a0 = *(const uint4*)(Ap + kk + kA);
    uint4 a1 = *(const uint4*)(Ap + kk + kA + 8);
    uint4 b0 = *(const uint4*)(Wp + kk + kB);
    uint4 b1 = *(const uint4*)(Wp + kk + kB + 8);
    v8u au = {a0.x, a0.y, a0.z, a0.w, a1.x, a1.y, a1.z, a1.w};
    v8u bu = {b0.x, b0.y, b0.z, b0.w, b1.x, b1.y, b1.z, b1.w};
    acc = __builtin_amdgcn_wmma_f32_16x16x32_bf16(
        false, __builtin_bit_cast(v16bf, au),
        false, __builtin_bit_cast(v16bf, bu),
        (short)0, acc, false, false);
  }

  const float bv   = bias ? bias[bcol] : 0.0f;
  const int   rofs = (lane < 16) ? 0 : 8;
#pragma unroll
  for (int r = 0; r < 8; ++r) {
    float v = acc[r] + bv;
    if (act == 1) v = (v >= 0.0f) ? v : 0.1f * v;           // LeakyReLU(0.1)
    size_t idx = (size_t)(rowbase + rofs + r) * ldc + bcol;
    if (Cf) Cf[idx] = v;
    if (Ch) Ch[idx] = f2bf(v);
  }
}

// =============================================================================
// GEMM #2 (fallback when ws too small for bf16 weights): A bf16, W fp32.
// W packed to bf16 with one v_perm_b32 per element pair (truncation).
// =============================================================================
__global__ void gemm_bw(const __bf16* __restrict__ A, const float* __restrict__ W,
                        const float* __restrict__ bias, float* __restrict__ Cf,
                        __bf16* __restrict__ Ch, int K, int ldc, int act) {
  const int lane    = threadIdx.x & 31;
  const int wave    = threadIdx.x >> 5;
  const int rowbase = blockIdx.x * 16;
  const int colbase = (blockIdx.y * 4 + wave) * 16;
  const int arow    = rowbase + (lane & 15);
  const int bcol    = colbase + (lane & 15);
  const int kA      = (lane < 16) ? 0 : 16;
  const int kB      = (lane < 16) ? 0 : 16;
  const __bf16* __restrict__ Ap = A + (size_t)arow * K;
  const float*  __restrict__ Wp = W + (size_t)bcol * K;

  v8f acc = {};
#pragma unroll 2
  for (int kk = 0; kk < K; kk += 32) {
    uint4 a0 = *(const uint4*)(Ap + kk + kA);
    uint4 a1 = *(const uint4*)(Ap + kk + kA + 8);
    uint4 w0 = *(const uint4*)(Wp + kk + kB);
    uint4 w1 = *(const uint4*)(Wp + kk + kB + 4);
    uint4 w2 = *(const uint4*)(Wp + kk + kB + 8);
    uint4 w3 = *(const uint4*)(Wp + kk + kB + 12);
    v8u au = {a0.x, a0.y, a0.z, a0.w, a1.x, a1.y, a1.z, a1.w};
    v8u bu = {pack_hi(w0.x, w0.y), pack_hi(w0.z, w0.w),
              pack_hi(w1.x, w1.y), pack_hi(w1.z, w1.w),
              pack_hi(w2.x, w2.y), pack_hi(w2.z, w2.w),
              pack_hi(w3.x, w3.y), pack_hi(w3.z, w3.w)};
    acc = __builtin_amdgcn_wmma_f32_16x16x32_bf16(
        false, __builtin_bit_cast(v16bf, au),
        false, __builtin_bit_cast(v16bf, bu),
        (short)0, acc, false, false);
  }

  const float bv   = bias ? bias[bcol] : 0.0f;
  const int   rofs = (lane < 16) ? 0 : 8;
#pragma unroll
  for (int r = 0; r < 8; ++r) {
    float v = acc[r] + bv;
    if (act == 1) v = (v >= 0.0f) ? v : 0.1f * v;
    size_t idx = (size_t)(rowbase + rofs + r) * ldc + bcol;
    if (Cf) Cf[idx] = v;
    if (Ch) Ch[idx] = f2bf(v);
  }
}

// ---------------------------------------------------------------------------
// Bulk fp32 -> bf16 converter (RNE), 8 elements/thread, b128 in / b128 out.
__global__ void cvt_bf16(const float* __restrict__ s, __bf16* __restrict__ d, long n) {
  long i = ((long)blockIdx.x * blockDim.x + threadIdx.x) * 8;
  if (i >= n) return;
  float4 f0 = *(const float4*)(s + i);
  float4 f1 = *(const float4*)(s + i + 4);
  __bf16 h[8] = {f2bf(f0.x), f2bf(f0.y), f2bf(f0.z), f2bf(f0.w),
                 f2bf(f1.x), f2bf(f1.y), f2bf(f1.z), f2bf(f1.w)};
  *(uint4*)(d + i) = *(const uint4*)h;
}

// ---------------------------------------------------------------------------
__global__ void init_state(float* __restrict__ h, __bf16* __restrict__ hb,
                           float* __restrict__ ctx, __bf16* __restrict__ xob,
                           float* __restrict__ cov, int* __restrict__ prev) {
  int i = blockIdx.x * blockDim.x + threadIdx.x;      // 512*256 = 131072
  if (i < B_ * H_)   { h[i] = 0.0f; hb[i] = (__bf16)0.0f; ctx[i] = 0.0f; }
  if (i < B_ * 2048) xob[i] = (__bf16)0.0f;           // [ctx0 | h0]
  if (i < B_ * L_)   cov[i] = 0.0f;
  if (i < B_)        prev[i] = 1;                     // BOS
}

// x_bf16 = [emb_table[prev], style_emb, ctx]
__global__ void build_x(const int* __restrict__ prev, const int* __restrict__ styles,
                        const float* __restrict__ emb, const float* __restrict__ style_tab,
                        const float* __restrict__ ctx, __bf16* __restrict__ xb) {
  const int b = blockIdx.x;
  for (int j = threadIdx.x; j < XDIM; j += blockDim.x) {
    float v;
    if (j < E_)            v = emb[(size_t)prev[b] * E_ + j];
    else if (j < E_ + S_)  v = style_tab[(size_t)styles[b] * S_ + (j - E_)];
    else                   v = ctx[(size_t)b * ENC_ + (j - E_ - S_)];
    xb[(size_t)b * XDIM + j] = f2bf(v);
  }
}

// GRU gate fusion: h' = (1-z)*n + z*h ; mirrors h' to bf16 and xo_b[:,1024:]
__global__ void gru_fuse(const float* __restrict__ gi, const float* __restrict__ gh,
                         float* __restrict__ h, __bf16* __restrict__ hb,
                         __bf16* __restrict__ xob) {
  int i = blockIdx.x * blockDim.x + threadIdx.x;
  if (i >= B_ * H_) return;
  int b = i / H_, j = i - b * H_;
  const float* gib = gi + (size_t)b * 3 * H_;
  const float* ghb = gh + (size_t)b * 3 * H_;
  float r  = 1.0f / (1.0f + __expf(-(gib[j] + ghb[j])));
  float z  = 1.0f / (1.0f + __expf(-(gib[H_ + j] + ghb[H_ + j])));
  float n  = tanhf(gib[2 * H_ + j] + r * ghb[2 * H_ + j]);
  float hn = (1.0f - z) * n + z * h[i];
  h[i]  = hn;
  hb[i] = f2bf(hn);
  xob[(size_t)b * 2048 + H_ + j] = f2bf(hn);
}

// Coverage attention: energies -> masked softmax -> ctx (f32 + bf16), cov += attn
__global__ void attn_step(const float* __restrict__ q, const float* __restrict__ keys,
                          const float* __restrict__ enc, const float* __restrict__ Wc,
                          const float* __restrict__ Wo, const int* __restrict__ enc_len,
                          float* __restrict__ cov, float* __restrict__ ctx,
                          __bf16* __restrict__ xob, float* __restrict__ attns, int t) {
  const int b    = blockIdx.x;
  const int tid  = threadIdx.x;
  const int lane = tid & 31;
  const int wave = tid >> 5;
  __shared__ float energy[L_];
  __shared__ float attnS[L_];
  __shared__ float red[L_];
  const int len = enc_len[b];
  const float* __restrict__ qp = q + (size_t)b * ATT_;

  for (int l = wave; l < L_; l += 8) {
    const float cv = cov[b * L_ + l];
    const float* __restrict__ kp = keys + (size_t)(b * L_ + l) * ATT_;
    float p = 0.0f;
    for (int a = lane; a < ATT_; a += 32)
      p += tanhf(qp[a] + kp[a] + cv * Wc[a]) * Wo[a];
    for (int off = 16; off > 0; off >>= 1) p += __shfl_down(p, off, 32);
    if (lane == 0) energy[l] = (l >= len) ? -1e10f : p;
  }
  __syncthreads();

  if (tid < L_) red[tid] = energy[tid];
  __syncthreads();
  for (int s = 32; s > 0; s >>= 1) {
    if (tid < s) red[tid] = fmaxf(red[tid], red[tid + s]);
    __syncthreads();
  }
  const float m = red[0];
  __syncthreads();
  if (tid < L_) { float e = __expf(energy[tid] - m); attnS[tid] = e; red[tid] = e; }
  __syncthreads();
  for (int s = 32; s > 0; s >>= 1) {
    if (tid < s) red[tid] += red[tid + s];
    __syncthreads();
  }
  const float inv = 1.0f / red[0];
  if (tid < L_) {
    float a = attnS[tid] * inv;
    attnS[tid] = a;
    cov[b * L_ + tid] += a;
    attns[((size_t)b * T_ + t) * L_ + tid] = a;
  }
  __syncthreads();

  for (int d = tid; d < ENC_; d += blockDim.x) {
    float s = 0.0f;
    const float* __restrict__ ep = enc + (size_t)b * L_ * ENC_ + d;
#pragma unroll 4
    for (int l = 0; l < L_; ++l) s += attnS[l] * ep[(size_t)l * ENC_];
    ctx[(size_t)b * ENC_ + d] = s;
    xob[(size_t)b * 2048 + d] = f2bf(s);
  }
}

// Argmax + log-softmax over V for one (b,t); feeds next-step prev token.
__global__ void argmax_logprob(const float* __restrict__ logits,
                               float* __restrict__ dlp, float* __restrict__ preds,
                               int* __restrict__ prev, int t) {
  const int b   = blockIdx.x;
  const int tid = threadIdx.x;
  const float* __restrict__ row = logits + ((size_t)b * T_ + t) * V_;
  __shared__ float smax[256];
  __shared__ int   sidx[256];
  __shared__ float ssum[256];

  float mv = -3.4e38f; int mi = V_;
  for (int v = tid; v < V_; v += 256) {
    float x = row[v];
    if (x > mv) { mv = x; mi = v; }
  }
  smax[tid] = mv; sidx[tid] = mi;
  __syncthreads();
  for (int s = 128; s > 0; s >>= 1) {
    if (tid < s) {
      float o = smax[tid + s]; int oi = sidx[tid + s];
      if (o > smax[tid] || (o == smax[tid] && oi < sidx[tid])) {
        smax[tid] = o; sidx[tid] = oi;
      }
    }
    __syncthreads();
  }
  const float gmax = smax[0];
  const int   gidx = sidx[0];

  float sum = 0.0f;
  for (int v = tid; v < V_; v += 256) sum += __expf(row[v] - gmax);
  ssum[tid] = sum;
  __syncthreads();
  for (int s = 128; s > 0; s >>= 1) {
    if (tid < s) ssum[tid] += ssum[tid + s];
    __syncthreads();
  }
  if (tid == 0) {
    float lse = gmax + __logf(ssum[0]);
    dlp[b * T_ + t]   = row[gidx] - lse;
    preds[b * T_ + t] = (float)gidx;
    prev[b] = gidx;
  }
}

// =============================================================================
extern "C" void kernel_launch(void* const* d_in, const int* in_sizes, int n_in,
                              void* d_out, int out_size, void* d_ws, size_t ws_size,
                              hipStream_t stream) {
  (void)in_sizes; (void)n_in; (void)out_size;
  const float* enc       = (const float*)d_in[0];
  const int*   enc_len   = (const int*)  d_in[1];
  const int*   styles    = (const int*)  d_in[2];
  const float* emb_table = (const float*)d_in[3];
  const float* style_tab = (const float*)d_in[4];
  const float* Wq        = (const float*)d_in[5];
  const float* bq        = (const float*)d_in[6];
  const float* Wk        = (const float*)d_in[7];
  const float* Wc        = (const float*)d_in[8];
  const float* Wo        = (const float*)d_in[9];
  const float* W_ih      = (const float*)d_in[10];
  const float* b_ih      = (const float*)d_in[11];
  const float* W_hh      = (const float*)d_in[12];
  const float* b_hh      = (const float*)d_in[13];
  const float* W1        = (const float*)d_in[14];
  const float* b1        = (const float*)d_in[15];
  const float* W2        = (const float*)d_in[16];
  const float* b2        = (const float*)d_in[17];

  float* out    = (float*)d_out;
  float* logits = out;                                   // [B,T,V]
  float* dlp    = out + (size_t)B_ * T_ * V_;            // [B,T]
  float* preds  = dlp + (size_t)B_ * T_;                 // [B,T] (float-encoded)
  float* attns  = preds + (size_t)B_ * T_;               // [B,T,L]

  // ---- workspace carve-up (256B aligned segments) ----
  char*  wsb = (char*)d_ws;
  size_t off = 0;
  auto alloc = [&](size_t bytes) -> void* {
    off = (off + 255) & ~(size_t)255;
    void* p = wsb + off; off += bytes; return p;
  };
  // base scratch (always used)
  float*  keys  = (float*) alloc((size_t)B_ * L_ * ATT_ * 4);
  __bf16* encb  = (__bf16*)alloc((size_t)B_ * L_ * ENC_ * 2);
  __bf16* xb    = (__bf16*)alloc((size_t)B_ * XDIM * 2);
  float*  gi    = (float*) alloc((size_t)B_ * 3 * H_ * 4);
  float*  gh    = (float*) alloc((size_t)B_ * 3 * H_ * 4);
  float*  h     = (float*) alloc((size_t)B_ * H_ * 4);
  __bf16* hb    = (__bf16*)alloc((size_t)B_ * H_ * 2);
  float*  ctx   = (float*) alloc((size_t)B_ * ENC_ * 4);
  __bf16* xob   = (__bf16*)alloc((size_t)B_ * 2048 * 2);
  float*  q     = (float*) alloc((size_t)B_ * ATT_ * 4);
  float*  cov   = (float*) alloc((size_t)B_ * L_ * 4);
  __bf16* obufb = (__bf16*)alloc((size_t)B_ * H_ * 2);
  int*    prev  = (int*)   alloc((size_t)B_ * 4);
  // bf16 weight pool (used only if ws_size is big enough)
  __bf16* W2b   = (__bf16*)alloc((size_t)V_ * H_ * 2);
  __bf16* Wihb  = (__bf16*)alloc((size_t)3 * H_ * XDIM * 2);
  __bf16* Whhb  = (__bf16*)alloc((size_t)3 * H_ * H_ * 2);
  __bf16* W1b   = (__bf16*)alloc((size_t)H_ * 2 * H_ * 2);
  __bf16* Wqb   = (__bf16*)alloc((size_t)ATT_ * H_ * 2);
  __bf16* Wkb   = (__bf16*)alloc((size_t)ATT_ * ENC_ * 2);
  const bool bigws = (ws_size >= off);

  auto cvt = [&](const float* s, __bf16* d, long n) {
    cvt_bf16<<<(unsigned)((n / 8 + 255) / 256), 256, 0, stream>>>(s, d, n);
  };

  // activations for the keys GEMM (enc is an input; convert once per call)
  cvt(enc, encb, (long)B_ * L_ * ENC_);
  if (bigws) {
    cvt(W2,   W2b,  (long)V_ * H_);
    cvt(W_ih, Wihb, (long)3 * H_ * XDIM);
    cvt(W_hh, Whhb, (long)3 * H_ * H_);
    cvt(W1,   W1b,  (long)H_ * 2 * H_);
    cvt(Wq,   Wqb,  (long)ATT_ * H_);
    cvt(Wk,   Wkb,  (long)ATT_ * ENC_);
  }
  init_state<<<512, 256, 0, stream>>>(h, hb, ctx, xob, cov, prev);

  // keys_proj = enc[4096,1024] @ Wk^T  (once per call)
  if (bigws)
    gemm_bb<<<dim3((B_ * L_) / 16, ATT_ / 64), 128, 0, stream>>>(
        encb, Wkb, nullptr, keys, nullptr, ENC_, ATT_, 0);
  else
    gemm_bw<<<dim3((B_ * L_) / 16, ATT_ / 64), 128, 0, stream>>>(
        encb, Wk, nullptr, keys, nullptr, ENC_, ATT_, 0);

  for (int t = 0; t < T_; ++t) {
    build_x<<<B_, 256, 0, stream>>>(prev, styles, emb_table, style_tab, ctx, xb);
    if (bigws) {
      gemm_bb<<<dim3(B_ / 16, (3 * H_) / 64), 128, 0, stream>>>(
          xb, Wihb, b_ih, gi, nullptr, XDIM, 3 * H_, 0);
      gemm_bb<<<dim3(B_ / 16, (3 * H_) / 64), 128, 0, stream>>>(
          hb, Whhb, b_hh, gh, nullptr, H_, 3 * H_, 0);
    } else {
      gemm_bw<<<dim3(B_ / 16, (3 * H_) / 64), 128, 0, stream>>>(
          xb, W_ih, b_ih, gi, nullptr, XDIM, 3 * H_, 0);
      gemm_bw<<<dim3(B_ / 16, (3 * H_) / 64), 128, 0, stream>>>(
          hb, W_hh, b_hh, gh, nullptr, H_, 3 * H_, 0);
    }
    gru_fuse<<<(B_ * H_) / 256, 256, 0, stream>>>(gi, gh, h, hb, xob);
    if (bigws)
      gemm_bb<<<dim3(B_ / 16, ATT_ / 64), 128, 0, stream>>>(
          hb, Wqb, bq, q, nullptr, H_, ATT_, 0);
    else
      gemm_bw<<<dim3(B_ / 16, ATT_ / 64), 128, 0, stream>>>(
          hb, Wq, bq, q, nullptr, H_, ATT_, 0);
    attn_step<<<B_, 256, 0, stream>>>(q, keys, enc, Wc, Wo, enc_len, cov, ctx, xob, attns, t);
    if (bigws) {
      gemm_bb<<<dim3(B_ / 16, H_ / 64), 128, 0, stream>>>(
          xob, W1b, b1, nullptr, obufb, 2 * H_, H_, 1);
      gemm_bb<<<dim3(B_ / 16, V_ / 64), 128, 0, stream>>>(
          obufb, W2b, b2, logits + (size_t)t * V_, nullptr, H_, T_ * V_, 0);
    } else {
      gemm_bw<<<dim3(B_ / 16, H_ / 64), 128, 0, stream>>>(
          xob, W1, b1, nullptr, obufb, 2 * H_, H_, 1);
      gemm_bw<<<dim3(B_ / 16, V_ / 64), 128, 0, stream>>>(
          obufb, W2, b2, logits + (size_t)t * V_, nullptr, H_, T_ * V_, 0);
    }
    argmax_logprob<<<B_, 256, 0, stream>>>(logits, dlp, preds, prev, t);
  }
}
